// SelfAttention_43284680409533
// MI455X (gfx1250) — compile-verified, matrix operand order checked by reading, not run on hardware
//
#include <hip/hip_runtime.h>
#include <hip/hip_bf16.h>

// ---------------------------------------------------------------------------
// Fused self-attention for MI455X (gfx1250, wave32, WMMA).
//
//   out[:,n] = gamma * ( sum_m  (V[:,m]/D[m]) * exp(S[m,n]) ) + x[:,n]
//   S[m,n]   = sum_q Q[q,m] K[q,n],    D[m] = sum_j exp(S[m,j])
//
// Pipeline (all on `stream`, workspace in d_ws):
//   1) k_proj_qk : Q,K = W*x+b  -> f16, stored [b][pixel][32] zero-padded K-dim
//   2) k_proj_v  : V  = Wv*x+bv -> f32, stored [b][c][pixel]
//   3) k_rowsum  : D[m] = rowsum(exp(Q^T K))      (f16 WMMA, S in registers)
//   4) k_scale_v : Vh = f16( V / D )
//   5) k_attn    : out = gamma * (Vh . exp(Q^T K)) + x   (chained WMMA, S/E
//                  tiles live entirely in VGPRs; E reuses the D-tile layout
//                  of the score WMMA directly as the B operand of the V WMMA)
//
// The 4096x4096 score matrix is never written to memory (saves 256 MB of HBM
// traffic @ 23.3 TB/s); recomputing it costs ~4 GFLOP of f16 WMMA == noise.
// ---------------------------------------------------------------------------

typedef __attribute__((ext_vector_type(16))) _Float16 v16h;
typedef __attribute__((ext_vector_type(8)))  _Float16 v8h;
typedef __attribute__((ext_vector_type(8)))  float    v8f;

#define BATCH 4
#define CCH   64
#define CQN   8
#define HWN   4096
#define KPAD  32   // channel (K) dimension of score WMMA, zero padded 8 -> 32

union V16HU { v16h v; v8h h[2]; };

// One WMMA f16 operand = 16 halves per lane = two aligned 16B chunks.
// A-operand (16x32, 16-bit): lane l (m = l&15, h = l>>4) needs
//   k = {8h..8h+7} and {16+8h..23+8h}  -> chunks {h, 2+h} of a 64B row.
// B-operand (32x16) has the identical per-lane pattern on the [n][32] layout.
static __device__ __forceinline__ v16h ld_op32(const _Float16* rowptr, int half) {
  V16HU u;
  u.h[0] = *(const v8h*)(rowptr + 8 * half);
  u.h[1] = *(const v8h*)(rowptr + 16 + 8 * half);
  return u.v;
}

static __device__ __forceinline__ v8f wmma_f16(v16h a, v16h b, v8f c) {
  // D = A x B + C, f32 accumulate
  return __builtin_amdgcn_wmma_f32_16x16x32_f16(false, a, false, b,
                                                (short)0, c, false, false);
}

// ---------------------------------------------------------------------------
// 1) Q/K projections. One thread per (b, n, q) with q in [0,32); q>=8 writes
//    the zero padding. Output layout [b][n][32] f16 (64B rows -> b128 gathers).
__global__ __launch_bounds__(256)
void k_proj_qk(const float* __restrict__ x,
               const float* __restrict__ Wq, const float* __restrict__ bq,
               const float* __restrict__ Wk, const float* __restrict__ bk,
               _Float16* __restrict__ Qp, _Float16* __restrict__ Kp) {
  int tid = blockIdx.x * blockDim.x + threadIdx.x;   // BATCH*HWN*KPAD threads
  int q = tid & (KPAD - 1);
  int n = (tid >> 5) & (HWN - 1);
  int b = tid >> 17;                                  // 4096*32 = 2^17
  size_t o = (size_t)tid;                             // == ((b*HWN)+n)*KPAD+q
  if (q >= CQN) { Qp[o] = (_Float16)0.f; Kp[o] = (_Float16)0.f; return; }
  const float* xc = x + (size_t)b * CCH * HWN + n;
  float sq = bq[q], sk = bk[q];
#pragma unroll 8
  for (int c = 0; c < CCH; ++c) {
    float xv = xc[(size_t)c * HWN];
    sq += Wq[q * CCH + c] * xv;
    sk += Wk[q * CCH + c] * xv;
  }
  Qp[o] = (_Float16)sq;
  Kp[o] = (_Float16)sk;
}

// ---------------------------------------------------------------------------
// 2) V projection (f32, layout [b][c][n]). One thread per output element.
__global__ __launch_bounds__(256)
void k_proj_v(const float* __restrict__ x,
              const float* __restrict__ Wv, const float* __restrict__ bv,
              float* __restrict__ V32) {
  int tid = blockIdx.x * blockDim.x + threadIdx.x;   // BATCH*CCH*HWN threads
  int n = tid & (HWN - 1);
  int c = (tid >> 12) & (CCH - 1);
  int b = tid >> 18;
  const float* xb = x + (size_t)b * CCH * HWN + n;
  float s = bv[c];
#pragma unroll 8
  for (int cc = 0; cc < CCH; ++cc) s += Wv[c * CCH + cc] * xb[(size_t)cc * HWN];
  V32[tid] = s;
}

// ---------------------------------------------------------------------------
// 3) D[m] = sum_j exp(S[m,j]).  One wave per 16-row m-strip; 256 j-tiles,
//    one V_WMMA_F32_16X16X32_F16 each; exp + accumulate in registers;
//    16-lane shuffle tree reduction at the end.
__global__ __launch_bounds__(256)
void k_rowsum(const _Float16* __restrict__ Qp, const _Float16* __restrict__ Kp,
              float* __restrict__ D) {
  int wave = (blockIdx.x * blockDim.x + threadIdx.x) >> 5;  // 0 .. BATCH*256-1
  int lane = threadIdx.x & 31;
  int b  = wave >> 8;
  int m0 = (wave & 255) * 16;
  int half = lane >> 4, nl = lane & 15;
  const _Float16* Qb = Qp + (size_t)b * HWN * KPAD;
  const _Float16* Kb = Kp + (size_t)b * HWN * KPAD;

  v16h a = ld_op32(Qb + (size_t)(m0 + nl) * KPAD, half);   // fixed A (Q rows)
  v8f acc = {};
  for (int j0 = 0; j0 < HWN; j0 += 16) {
    v16h bm = ld_op32(Kb + (size_t)(j0 + nl) * KPAD, half);
    v8f s = {};
    s = wmma_f16(a, bm, s);
#pragma unroll
    for (int r = 0; r < 8; ++r) acc[r] += __expf(s[r]);
  }
  // D-tile: lane (h,nl), reg r  <->  row m0+8h+r, col nl. Reduce over nl.
#pragma unroll
  for (int r = 0; r < 8; ++r) {
    float v = acc[r];
    v += __shfl_xor(v, 1, 32);
    v += __shfl_xor(v, 2, 32);
    v += __shfl_xor(v, 4, 32);
    v += __shfl_xor(v, 8, 32);
    if (nl == 0) D[b * HWN + m0 + 8 * half + r] = v;
  }
}

// ---------------------------------------------------------------------------
// 4) Vh = f16( V32 / D ), layout [b][c][m].
__global__ __launch_bounds__(256)
void k_scale_v(const float* __restrict__ V32, const float* __restrict__ D,
               _Float16* __restrict__ Vh) {
  int tid = blockIdx.x * blockDim.x + threadIdx.x;
  int m = tid & (HWN - 1);
  int b = tid >> 18;
  Vh[tid] = (_Float16)(V32[tid] / D[b * HWN + m]);
}

// ---------------------------------------------------------------------------
// 5) out[c, n0..n0+15] = gamma * sum_m Vh[c,m] * exp(S[m,n]) + x.
//    One wave per 16-column n-tile. Per 32 m-rows: 2 score WMMAs -> exp ->
//    pack as f16 B operand (exact register-layout match, zero shuffles) ->
//    4 accumulate WMMAs covering c = 0..63.
__global__ __launch_bounds__(256)
void k_attn(const _Float16* __restrict__ Qp, const _Float16* __restrict__ Kp,
            const _Float16* __restrict__ Vh, const float* __restrict__ x,
            const float* __restrict__ gamma, float* __restrict__ out) {
  int wave = (blockIdx.x * blockDim.x + threadIdx.x) >> 5;  // 0 .. BATCH*256-1
  int lane = threadIdx.x & 31;
  int b  = wave >> 8;
  int n0 = (wave & 255) * 16;
  int half = lane >> 4, nl = lane & 15;

  const _Float16* Qb = Qp + (size_t)b * HWN * KPAD;
  const _Float16* Kb = Kp + (size_t)b * HWN * KPAD;
  const _Float16* Vb = Vh + (size_t)b * CCH * HWN;

  v16h bK = ld_op32(Kb + (size_t)(n0 + nl) * KPAD, half);  // fixed B (K cols)

  v8f acc[4] = {{}, {}, {}, {}};   // out tiles c = [0,16), [16,32), [32,48), [48,64)

  for (int m0 = 0; m0 < HWN; m0 += 32) {
    // scores S[m0..m0+31][n0..n0+15], K-dim = q (8, zero-padded to 32)
    v16h aQ0 = ld_op32(Qb + (size_t)(m0 + nl) * KPAD, half);
    v16h aQ1 = ld_op32(Qb + (size_t)(m0 + 16 + nl) * KPAD, half);
    if (m0 + 32 < HWN)
      __builtin_prefetch(Qb + (size_t)(m0 + 32 + nl) * KPAD, 0, 3);
    v8f s0 = {}, s1 = {};
    s0 = wmma_f16(aQ0, bK, s0);
    s1 = wmma_f16(aQ1, bK, s1);

    // E = exp(S) packed as the 32x16 f16 B operand. D-tile reg r of lane
    // (h,nl) is row 8h+r -> exactly B-operand element j=r (k=8h+r) /
    // j=8+r (k=16+8h+r). Pure per-lane repack, no cross-lane traffic.
    V16HU e;
#pragma unroll
    for (int r = 0; r < 8; ++r) {
      e.v[r]     = (_Float16)__expf(s0[r]);
      e.v[8 + r] = (_Float16)__expf(s1[r]);
    }

    // acc[cb] += Vh[16cb .. 16cb+15][m0 .. m0+31] * E
#pragma unroll
    for (int cb = 0; cb < 4; ++cb) {
      v16h aV = ld_op32(Vb + (size_t)(cb * 16 + nl) * HWN + m0, half);
      acc[cb] = wmma_f16(aV, e.v, acc[cb]);
    }
  }

  // Epilogue: out = gamma * acc + x.  D-tile: reg r of lane (h,nl) is
  // row c = 16cb + 8h + r, col n = n0 + nl.
  float g = gamma[0];
  const float* xb = x + (size_t)b * CCH * HWN;
  float* ob = out + (size_t)b * CCH * HWN;
#pragma unroll
  for (int cb = 0; cb < 4; ++cb) {
#pragma unroll
    for (int r = 0; r < 8; ++r) {
      int c = 16 * cb + 8 * half + r;
      size_t idx = (size_t)c * HWN + n0 + nl;
      ob[idx] = g * acc[cb][r] + xb[idx];
    }
  }
}

// ---------------------------------------------------------------------------
extern "C" void kernel_launch(void* const* d_in, const int* in_sizes, int n_in,
                              void* d_out, int out_size, void* d_ws, size_t ws_size,
                              hipStream_t stream) {
  const float* x     = (const float*)d_in[0];
  const float* Wq    = (const float*)d_in[1];
  const float* bq    = (const float*)d_in[2];
  const float* Wk    = (const float*)d_in[3];
  const float* bk    = (const float*)d_in[4];
  const float* Wv    = (const float*)d_in[5];
  const float* bv    = (const float*)d_in[6];
  const float* gamma = (const float*)d_in[7];
  float* out = (float*)d_out;

  // Workspace layout (~8.1 MB total)
  char* ws = (char*)d_ws;
  _Float16* Qp  = (_Float16*)(ws);                          // 4*4096*32*2 = 1 MB
  _Float16* Kp  = (_Float16*)(ws + (1u << 20));             // 1 MB
  float*    V32 = (float*)   (ws + (2u << 20));             // 4*64*4096*4 = 4 MB
  _Float16* Vh  = (_Float16*)(ws + (6u << 20));             // 2 MB
  float*    Dd  = (float*)   (ws + (8u << 20));             // 4*4096*4 = 64 KB

  const int TB = 256;
  // 1) Q/K projection: BATCH*HWN*KPAD threads
  k_proj_qk<<<(BATCH * HWN * KPAD) / TB, TB, 0, stream>>>(x, Wq, bq, Wk, bk, Qp, Kp);
  // 2) V projection: BATCH*CCH*HWN threads
  k_proj_v<<<(BATCH * CCH * HWN) / TB, TB, 0, stream>>>(x, Wv, bv, V32);
  // 3) softmax denominators: BATCH*256 waves
  k_rowsum<<<(BATCH * 256 * 32) / TB, TB, 0, stream>>>(Qp, Kp, Dd);
  // 4) scale V by 1/D, convert to f16
  k_scale_v<<<(BATCH * CCH * HWN) / TB, TB, 0, stream>>>(V32, Dd, Vh);
  // 5) fused attention + residual: BATCH*256 waves
  k_attn<<<(BATCH * 256 * 32) / TB, TB, 0, stream>>>(Qp, Kp, Vh, x, gamma, out);
}
